// RGCNEncoderDecoder_31336081392095
// MI455X (gfx1250) — matrix operand order repaired and checked by reading, not compiled, not run on hardware
//
#include <hip/hip_runtime.h>

typedef __bf16 bf16x16 __attribute__((ext_vector_type(16)));
typedef __bf16 bf16x2  __attribute__((ext_vector_type(2)));
typedef float  f32x8   __attribute__((ext_vector_type(8)));
typedef float  f32x4   __attribute__((ext_vector_type(4)));

#define NGRAPH 262144
#define NTILES (NGRAPH / 16)
#define NBASIS 10

__device__ __forceinline__ unsigned short f2bf(float f) {
  union { float f; unsigned int u; } c; c.f = f;
  unsigned int u = c.u + 0x7FFFu + ((c.u >> 16) & 1u);   // round-to-nearest-even
  return (unsigned short)(u >> 16);
}

// Pack two f32 -> packed bf16x2 (low = a, high = b).
// Prefer HW packed convert; else round-half-up + single v_perm_b32 byte-select.
__device__ __forceinline__ unsigned int pkbf(float a, float b) {
#if __has_builtin(__builtin_amdgcn_cvt_pk_bf16_f32)
  union { bf16x2 v; unsigned int u; } c;
  c.v = __builtin_amdgcn_cvt_pk_bf16_f32(a, b);
  return c.u;
#elif __has_builtin(__builtin_amdgcn_perm)
  union { float f; unsigned int u; } ca, cb;
  ca.f = a; cb.f = b;
  unsigned int ra = ca.u + 0x8000u;   // round-half-up into bf16 mantissa
  unsigned int rb = cb.u + 0x8000u;
  // D.byte0 = ra.byte2, D.byte1 = ra.byte3, D.byte2 = rb.byte2, D.byte3 = rb.byte3
  return __builtin_amdgcn_perm(rb, ra, 0x07060302u);
#else
  return (unsigned int)f2bf(a) | ((unsigned int)f2bf(b) << 16);
#endif
}

// ---------------------------------------------------------------------------
// Kernel 1: build the 4 combined 128x128 weight matrices (root, W_msg[1..3])
// in bf16, pre-swizzled into the WMMA B-operand per-lane fragment layout:
//   fragment (w = n/16, fi = mat*4 + k/32), lane = (k%32)/16*16 + n%16,
//   element j = k%16  holds  W[k][n].
// ---------------------------------------------------------------------------
__global__ __launch_bounds__(256) void build_weights(
    const float* __restrict__ basis, const float* __restrict__ att,
    const float* __restrict__ root,  const int* __restrict__ edge_type,
    unsigned short* __restrict__ wsb) {
  int tid = blockIdx.x * 256 + threadIdx.x;
  for (int e = tid; e < 4 * 128 * 128; e += 64 * 256) {
    int m   = e >> 14;        // 0 = root, 1..3 = message weight for node m
    int idx = e & 16383;      // k*128 + n
    float val;
    if (m == 0) {
      val = root[idx];
    } else {
      int rel = edge_type[3 - m];   // node m receives along edge (3-m)
      const float* ar = att + rel * NBASIS;
      val = 0.f;
      #pragma unroll
      for (int b = 0; b < NBASIS; ++b) val += ar[b] * basis[(b << 14) + idx];
    }
    int k = idx >> 7, n = idx & 127;
    int w = n >> 4, nl = n & 15;
    int kc = k >> 5, r = k & 31;
    int half = r >> 4, j = r & 15;
    int lane = (half << 4) | nl;
    int fi = m * 4 + kc;
    wsb[(((w * 16 + fi) * 32 + lane) << 4) + j] = f2bf(val);
  }
}

// ---------------------------------------------------------------------------
// Kernel 2: per 16-graph tile, compute h = relu([x_self,x_src]@[root;W]+bias),
// pool over the 4 nodes, cosine with target. 8 waves = 8 N-tiles of 16 cols.
// ---------------------------------------------------------------------------
__global__ __launch_bounds__(256, 4) void rgcn_main(
    const float* __restrict__ x, const float* __restrict__ bias,
    const float* __restrict__ target, const unsigned short* __restrict__ wsb,
    float* __restrict__ out) {
  __shared__ __align__(16) unsigned short xbf[64 * 132]; // stride 132 vs banks
  __shared__ __align__(16) float pooled[16 * 128];

  const int tid  = threadIdx.x;
  const int w    = tid >> 5;      // wave id = N-tile (cols 16w..16w+15)
  const int lane = tid & 31;
  const int mloc = lane & 15;     // M row (graph within tile)
  const int half = lane >> 4;

  union Frag { bf16x16 v; uint4 q[2]; uint2 d[4]; };

  // Load this wave's 16 B-operand fragments (root kc0..3, W1, W2, W3) from L2.
  Frag wf[16];
  {
    const unsigned short* wp = wsb + (((size_t)w * 16 * 32 + lane) << 4);
    #pragma unroll
    for (int fi = 0; fi < 16; ++fi) {
      const uint4* g = (const uint4*)(wp + fi * 512);
      wf[fi].q[0] = g[0];
      wf[fi].q[1] = g[1];
    }
  }
  const float bval = bias[(w << 4) + mloc];
  const int srcnode[4] = {0, 2, 3, 0};   // node i receives message from srcnode[i]

  for (int tile = blockIdx.x; tile < NTILES; tile += gridDim.x) {
    const int g0 = tile << 4;

    // Stage 64 x-rows (4 nodes x 16 graphs) as bf16 into LDS:
    // float4 NT global loads -> packed bf16 -> ds_store_b64.
    const f32x4* xs4 = (const f32x4*)(x + ((size_t)g0 << 9));
    #pragma unroll
    for (int it = 0; it < 8; ++it) {
      int idx4 = it * 256 + tid;              // float4 index within 64x128 tile
      f32x4 v = __builtin_nontemporal_load(xs4 + idx4);
      int row = idx4 >> 5;                    // 32 float4 per row
      int c4  = (idx4 & 31) << 2;             // col (multiple of 4)
      uint2 s;
      s.x = pkbf(v[0], v[1]);
      s.y = pkbf(v[2], v[3]);
      *(uint2*)(xbf + row * 132 + c4) = s;
    }
    // Prefetch next tile's streams while this tile computes.
    {
      int ntile = tile + (int)gridDim.x;
      if (ntile < NTILES) {
        const char* nx = (const char*)(x + ((size_t)ntile << 13));
        __builtin_prefetch(nx + tid * 128, 0, 0);          // 32 KB of x
        const char* nt = (const char*)(target + ((size_t)ntile << 11));
        __builtin_prefetch(nt + tid * 32, 0, 0);           // 8 KB of target
      }
    }
    __syncthreads();

    f32x8 pool = {0.f, 0.f, 0.f, 0.f, 0.f, 0.f, 0.f, 0.f};
    #pragma unroll
    for (int i = 0; i < 4; ++i) {
      f32x8 acc = {0.f, 0.f, 0.f, 0.f, 0.f, 0.f, 0.f, 0.f};
      const int nkc = (i == 0) ? 4 : 8;   // node 0 has no incoming message
      #pragma unroll
      for (int kc = 0; kc < nkc; ++kc) {
        // Gather A fragment (contiguous K order, matching B):
        //  a[j] : K = 32*kc' + 16*half + j
        int row  = (kc < 4) ? (4 * mloc + i) : (4 * mloc + srcnode[i]);
        int cb   = ((kc & 3) << 5) + (half << 4);
        int base = row * 132 + cb;
        Frag A;
        const uint2* p  = (const uint2*)(xbf + base);
        A.d[0] = p[0];  A.d[1] = p[1];
        const uint2* p2 = (const uint2*)(xbf + base + 8);
        A.d[2] = p2[0]; A.d[3] = p2[1];
        int fidx = (kc < 4) ? kc : (i * 4 + (kc - 4));
        acc = __builtin_amdgcn_wmma_f32_16x16x32_bf16(
            false, A.v, false, wf[fidx].v, (short)0, acc, false, false);
      }
      #pragma unroll
      for (int r = 0; r < 8; ++r) pool[r] += fmaxf(acc[r] + bval, 0.f);
    }
    __syncthreads();

    // Spill pooled tile (C layout: VGPR r -> row r + 8*half, col = lane%16).
    {
      int mb = half << 3;
      int colp = (w << 4) + mloc;
      #pragma unroll
      for (int r = 0; r < 8; ++r) pooled[(mb + r) * 128 + colp] = pool[r];
    }
    __syncthreads();

    // Cosine: 16 threads per graph, each owns a contiguous 8-col chunk,
    // then shuffle-reduce within the 16-lane group.
    {
      int g = tid >> 4, jj = tid & 15;
      const f32x4* tg4 = (const f32x4*)(target + (((size_t)(g0 + g)) << 7) + (jj << 3));
      f32x4 t0 = __builtin_nontemporal_load(tg4);
      f32x4 t1 = __builtin_nontemporal_load(tg4 + 1);
      const f32x4* pp = (const f32x4*)(pooled + g * 128 + (jj << 3));
      f32x4 p0 = pp[0];
      f32x4 p1 = pp[1];
      float dot = 0.f, ss = 0.f, st = 0.f;
      #pragma unroll
      for (int c = 0; c < 4; ++c) {
        dot += p0[c] * t0[c] + p1[c] * t1[c];
        ss  += p0[c] * p0[c] + p1[c] * p1[c];
        st  += t0[c] * t0[c] + t1[c] * t1[c];
      }
      #pragma unroll
      for (int off = 8; off >= 1; off >>= 1) {
        dot += __shfl_xor(dot, off, 16);
        ss  += __shfl_xor(ss,  off, 16);
        st  += __shfl_xor(st,  off, 16);
      }
      if (jj == 0)
        out[g0 + g] = dot / (fmaxf(sqrtf(ss), 1e-8f) * fmaxf(sqrtf(st), 1e-8f));
    }
  }
}

extern "C" void kernel_launch(void* const* d_in, const int* in_sizes, int n_in,
                              void* d_out, int out_size, void* d_ws, size_t ws_size,
                              hipStream_t stream) {
  (void)in_sizes; (void)n_in; (void)out_size; (void)ws_size;
  const float* x         = (const float*)d_in[0];
  const float* basis     = (const float*)d_in[1];
  const float* att       = (const float*)d_in[2];
  const float* root      = (const float*)d_in[3];
  const float* bias      = (const float*)d_in[4];
  const float* target    = (const float*)d_in[5];
  const int*   edge_type = (const int*)d_in[7];
  unsigned short* wsb = (unsigned short*)d_ws;   // 128 KB of bf16 fragments
  float* out = (float*)d_out;

  build_weights<<<64, 256, 0, stream>>>(basis, att, root, edge_type, wsb);
  rgcn_main<<<2048, 256, 0, stream>>>(x, bias, target, wsb, out);
}
